// CrossTransFormer_86526411145604
// MI455X (gfx1250) — matrix operand order repaired and checked by reading, not compiled, not run on hardware
//
#include <hip/hip_runtime.h>
#include <hip/hip_bf16.h>

#define C_DIM 1024
#define HEADS 8
#define DH 128
#define BATCH 4
#define P_DIM 2048
#define EPS_BN 1e-5f
#define SLOPE 0.1f

typedef __attribute__((ext_vector_type(16))) __bf16 v16bf;
typedef __attribute__((ext_vector_type(8)))  __bf16 v8bf;
typedef __attribute__((ext_vector_type(4)))  __bf16 v4bf;
typedef __attribute__((ext_vector_type(8)))  float  v8f;

// ---------------------------------------------------------------------------
// Fragment gathers matching CDNA5 WMMA 16x16x32 bf16 per-lane layouts.
// A (16x32, MxK): lane = row (lane&15); K chunks [hi*8, hi*8+8) and [16+hi*8, ...)
// B (32x16, KxN): lane = col (lane&15); K chunks [hi*8, hi*8+8) and [16+hi*8, ...)
// ---------------------------------------------------------------------------
__device__ __forceinline__ v16bf load_a_frag(const __bf16* rowPtr, int hi) {
  v8bf a0 = *reinterpret_cast<const v8bf*>(rowPtr + hi * 8);
  v8bf a1 = *reinterpret_cast<const v8bf*>(rowPtr + 16 + hi * 8);
  v16bf a;
#pragma unroll
  for (int e = 0; e < 8; ++e) { a[e] = a0[e]; a[8 + e] = a1[e]; }
  return a;
}

__device__ __forceinline__ v16bf load_b_frag(const __bf16* colPtr, int hi) {
  v8bf b0 = *reinterpret_cast<const v8bf*>(colPtr + hi * 8);
  v8bf b1 = *reinterpret_cast<const v8bf*>(colPtr + 16 + hi * 8);
  v16bf b;
#pragma unroll
  for (int e = 0; e < 8; ++e) { b[e] = b0[e]; b[8 + e] = b1[e]; }
  return b;
}

// ---------------------------------------------------------------------------
// Fold BatchNorm eval stats into per-channel scale/shift (3 sets of C each).
// ---------------------------------------------------------------------------
__global__ void bn_coeffs_kernel(
    const float* g1, const float* b1, const float* m1, const float* v1,
    const float* g2, const float* b2, const float* m2, const float* v2,
    const float* g3, const float* b3, const float* m3, const float* v3,
    float* sc, float* sh) {
  int i = blockIdx.x * blockDim.x + threadIdx.x;
  if (i >= 3 * C_DIM) return;
  int s = i / C_DIM, c = i - s * C_DIM;
  const float* G = (s == 0) ? g1 : (s == 1) ? g2 : g3;
  const float* B = (s == 0) ? b1 : (s == 1) ? b2 : b3;
  const float* M = (s == 0) ? m1 : (s == 1) ? m2 : m3;
  const float* V = (s == 0) ? v1 : (s == 1) ? v2 : v3;
  float scale = G[c] * rsqrtf(V[c] + EPS_BN);
  sc[i] = scale;
  sh[i] = B[c] - M[c] * scale;
}

// ---------------------------------------------------------------------------
// Y = LeakyReLU(scale * (W @ X) + shift), output bf16.
// Block tile 128(M=out ch) x 128(N=pos), K step 32. 8 waves in 4x2 layout,
// each wave 32x64 (2x4 WMMA tiles). TRANS: 0 -> [b][C][P], 1 -> [b][P][C].
// Output staged through LDS so global stores are coalesced b128.
// ---------------------------------------------------------------------------
#define LDA 40       // padded bf16 row stride for K tiles (80B, 16B aligned)
#define LDS_STG 136  // padded bf16 row stride for output staging (272B)

template <int TRANS>
__global__ __launch_bounds__(256) void qkv_gemm_bn_kernel(
    const float* __restrict__ W, const float* __restrict__ X,
    const float* __restrict__ sc, const float* __restrict__ sh,
    unsigned short* __restrict__ OutRaw) {
  __shared__ __bf16 smem[128 * LDS_STG];  // K-loop: lsA | lsB ; epilogue: staging
  __bf16* lsA = smem;                 // W tile [m][k], 128*40
  __bf16* lsB = smem + 128 * LDA;     // X tile [n][k] (transposed), 128*40
  __bf16* Out = reinterpret_cast<__bf16*>(OutRaw);

  const int t = threadIdx.x;
  const int lane = t & 31, wid = t >> 5;
  const int hi = lane >> 4, nl = lane & 15;
  const int wm = (wid & 3) * 32;
  const int wn = (wid >> 2) * 64;
  const int n0 = blockIdx.x * 128;
  const int m0 = blockIdx.y * 128;
  const int bz = blockIdx.z;

  v8f acc[2][4];
#pragma unroll
  for (int mi = 0; mi < 2; ++mi)
#pragma unroll
    for (int ni = 0; ni < 4; ++ni) acc[mi][ni] = {};

  for (int k0 = 0; k0 < C_DIM; k0 += 32) {
    // W tile: 128x32, float4 loads, packed b64 LDS stores.
#pragma unroll
    for (int e = 0; e < 4; ++e) {
      int idx = e * 256 + t;            // 1024 chunks of 4 floats
      int r = idx >> 3, c4 = (idx & 7) * 4;
      float4 f = *reinterpret_cast<const float4*>(
          &W[(size_t)(m0 + r) * C_DIM + k0 + c4]);
      v4bf p;
      p[0] = (__bf16)f.x; p[1] = (__bf16)f.y;
      p[2] = (__bf16)f.z; p[3] = (__bf16)f.w;
      *reinterpret_cast<v4bf*>(&lsA[r * LDA + c4]) = p;
    }
    // X tile: 32x128, float4 loads, transposed scalar LDS stores.
#pragma unroll
    for (int e = 0; e < 4; ++e) {
      int idx = e * 256 + t;            // 1024 chunks of 4 floats
      int kk = idx >> 5, n4 = (idx & 31) * 4;
      float4 f = *reinterpret_cast<const float4*>(
          &X[((size_t)bz * C_DIM + k0 + kk) * P_DIM + n0 + n4]);
      lsB[(n4 + 0) * LDA + kk] = (__bf16)f.x;
      lsB[(n4 + 1) * LDA + kk] = (__bf16)f.y;
      lsB[(n4 + 2) * LDA + kk] = (__bf16)f.z;
      lsB[(n4 + 3) * LDA + kk] = (__bf16)f.w;
    }
    if (k0 + 32 < C_DIM) {  // prefetch next tiles into cache
      __builtin_prefetch(&W[(size_t)(m0 + (t >> 1)) * C_DIM + k0 + 32], 0, 0);
      __builtin_prefetch(
          &X[((size_t)bz * C_DIM + k0 + 32 + (t >> 3)) * P_DIM + n0 + (t & 7) * 16],
          0, 0);
    }
    __syncthreads();

    v16bf aF[2], bF[4];
#pragma unroll
    for (int mi = 0; mi < 2; ++mi)
      aF[mi] = load_a_frag(&lsA[(wm + mi * 16 + nl) * LDA], hi);
#pragma unroll
    for (int ni = 0; ni < 4; ++ni)
      bF[ni] = load_b_frag(&lsB[(wn + ni * 16 + nl) * LDA], hi);
#pragma unroll
    for (int mi = 0; mi < 2; ++mi)
#pragma unroll
      for (int ni = 0; ni < 4; ++ni)
        acc[mi][ni] = __builtin_amdgcn_wmma_f32_16x16x32_bf16(
            false, aF[mi], false, bF[ni], (short)0, acc[mi][ni], false, false);
    __syncthreads();
  }

  // Epilogue: BN affine + LeakyReLU -> bf16, staged via LDS.
#pragma unroll
  for (int mi = 0; mi < 2; ++mi)
#pragma unroll
    for (int r = 0; r < 8; ++r) {
      int ol = wm + mi * 16 + r + 8 * hi;
      float scv = sc[m0 + ol], shv = sh[m0 + ol];
#pragma unroll
      for (int ni = 0; ni < 4; ++ni) {
        int pl = wn + ni * 16 + nl;
        float y = fmaf(acc[mi][ni][r], scv, shv);
        y = (y > 0.f) ? y : SLOPE * y;
        int idx = TRANS ? (pl * LDS_STG + ol) : (ol * LDS_STG + pl);
        smem[idx] = (__bf16)y;
      }
    }
  __syncthreads();

  // Cooperative coalesced store: 128 rows x 16 chunks of 8 bf16 (16B).
#pragma unroll
  for (int e = 0; e < 8; ++e) {
    int chunk = e * 256 + t;
    int row = chunk >> 4, ch = chunk & 15;
    uint4 v = *reinterpret_cast<const uint4*>(&smem[row * LDS_STG + ch * 8]);
    size_t dst = TRANS
        ? (((size_t)bz * P_DIM + n0 + row) * C_DIM + m0 + ch * 8)
        : (((size_t)bz * C_DIM + m0 + row) * P_DIM + n0 + ch * 8);
    *reinterpret_cast<uint4*>(&Out[dst]) = v;
  }
}

// ---------------------------------------------------------------------------
// Flash-style fused attention core.
//   S[i,j] = sum_c K[c,i] V[c,j] / sqrt(DH);  P = softmax_j(S)
//   out[c,i] = sum_j Q[c,j] P[i,j]            (computed as Q · P^T)
// Layouts: Q [b][C][P] (contig in j), Kt/Vt [b][P][C] (contig in c).
// Fragment loads are batched in groups of 4 so one loadcnt wait covers
// four v_wmma issues instead of fencing each one.
// ---------------------------------------------------------------------------
#define LDP 136  // padded bf16 row stride for P staging (272B)

__global__ __launch_bounds__(256) void xattn_kernel(
    const unsigned short* __restrict__ QbRaw,
    const unsigned short* __restrict__ KtRaw,
    const unsigned short* __restrict__ VtRaw,
    float* __restrict__ Out) {
  __shared__ __bf16 lsP[128 * LDP];   // P tile, [i_local][j_local]
  __shared__ float lsStat[128];       // per-row broadcast (corr, then 1/l)
  const __bf16* Qb = reinterpret_cast<const __bf16*>(QbRaw);
  const __bf16* Kt = reinterpret_cast<const __bf16*>(KtRaw);
  const __bf16* Vt = reinterpret_cast<const __bf16*>(VtRaw);

  const int t = threadIdx.x;
  const int lane = t & 31, w = t >> 5;
  const int hi = lane >> 4, nl = lane & 15;
  const int i0 = blockIdx.x * 128 + w * 16;  // this wave's 16 attention rows
  const int h = blockIdx.y;
  const int bz = blockIdx.z;
  const float scl = 0.08838834764831845f;  // 1/sqrt(128)

  // Hoist the 4 K fragments (rows i, K-dim c=128) for the whole j loop.
  const __bf16* Krow = Kt + ((size_t)bz * P_DIM + i0 + nl) * C_DIM;
  v16bf kF[4];
#pragma unroll
  for (int kk = 0; kk < 4; ++kk) kF[kk] = load_a_frag(Krow + kk * 32, hi);

  v8f accO[8];        // M = c (8 tiles of 16), N = i ; lane's i = i0 + nl
  float mrow[8], lrow[8];  // stats for rows i = i0 + 8*hi + r (S C-layout)
#pragma unroll
  for (int q = 0; q < 8; ++q) { accO[q] = {}; mrow[q] = -3.0e38f; lrow[q] = 0.f; }

  for (int j0 = 0; j0 < P_DIM; j0 += 128) {
    if (j0 + 128 < P_DIM) {  // prefetch next V/Q tiles
      __builtin_prefetch(
          Vt + ((size_t)bz * P_DIM + j0 + 128 + lane * 4) * C_DIM, 0, 0);
      __builtin_prefetch(
          Qb + ((size_t)bz * C_DIM + h * DH + lane * 4) * P_DIM + j0 + 128, 0, 0);
    }

    // ---- S tile: 16(i) x 128(j), K-dim = 128 channels ----
    v8f accS[8];
#pragma unroll
    for (int q = 0; q < 8; ++q) accS[q] = {};
#pragma unroll
    for (int kk = 0; kk < 4; ++kk) {
      const __bf16* vbase =
          Vt + ((size_t)bz * P_DIM + j0 + nl) * C_DIM + kk * 32;
#pragma unroll
      for (int g = 0; g < 2; ++g) {
        v16bf bF[4];
#pragma unroll
        for (int u = 0; u < 4; ++u)
          bF[u] = load_b_frag(vbase + (size_t)(g * 4 + u) * 16 * C_DIM, hi);
#pragma unroll
        for (int u = 0; u < 4; ++u)
          accS[g * 4 + u] = __builtin_amdgcn_wmma_f32_16x16x32_bf16(
              false, kF[kk], false, bF[u], (short)0, accS[g * 4 + u], false,
              false);
      }
    }

    // ---- online softmax per row (S C-layout: row i = i0 + r + 8*hi) ----
    float corrA[8];
#pragma unroll
    for (int r = 0; r < 8; ++r) {
      float mx = -3.0e38f;
#pragma unroll
      for (int ni = 0; ni < 8; ++ni) mx = fmaxf(mx, accS[ni][r]);
      mx *= scl;
#pragma unroll
      for (int msk = 8; msk >= 1; msk >>= 1)
        mx = fmaxf(mx, __shfl_xor(mx, msk));
      float mnew = fmaxf(mrow[r], mx);
      corrA[r] = __expf(mrow[r] - mnew);
      mrow[r] = mnew;
      float sum = 0.f;
#pragma unroll
      for (int ni = 0; ni < 8; ++ni) {
        float p = __expf(fmaf(accS[ni][r], scl, -mnew));
        accS[ni][r] = p;
        sum += p;
      }
#pragma unroll
      for (int msk = 8; msk >= 1; msk >>= 1) sum += __shfl_xor(sum, msk);
      lrow[r] = lrow[r] * corrA[r] + sum;
    }

    // ---- stage P to LDS + broadcast corr per i-row (wave-local) ----
    if (nl == 0) {
#pragma unroll
      for (int r = 0; r < 8; ++r) lsStat[w * 16 + 8 * hi + r] = corrA[r];
    }
#pragma unroll
    for (int ni = 0; ni < 8; ++ni)
#pragma unroll
      for (int r = 0; r < 8; ++r)
        lsP[(w * 16 + r + 8 * hi) * LDP + ni * 16 + nl] = (__bf16)accS[ni][r];
    asm volatile("s_wait_dscnt 0" ::: "memory");

    float corrMy = lsStat[w * 16 + nl];  // lane's i = i0 + nl
#pragma unroll
    for (int ci = 0; ci < 8; ++ci)
#pragma unroll
      for (int r = 0; r < 8; ++r) accO[ci][r] *= corrMy;

    // ---- accO += Q · P^T : A = Q rows (M=c), B = P cols (N=i), K=j ----
#pragma unroll
    for (int jk = 0; jk < 4; ++jk) {
      v16bf pF = load_b_frag(&lsP[(w * 16 + nl) * LDP + jk * 32], hi);
      const __bf16* qbase =
          Qb + ((size_t)bz * C_DIM + h * DH + nl) * P_DIM + j0 + jk * 32;
#pragma unroll
      for (int g = 0; g < 2; ++g) {
        v16bf qF[4];
#pragma unroll
        for (int u = 0; u < 4; ++u)
          qF[u] = load_a_frag(qbase + (size_t)(g * 4 + u) * 16 * P_DIM, hi);
#pragma unroll
        for (int u = 0; u < 4; ++u)
          accO[g * 4 + u] = __builtin_amdgcn_wmma_f32_16x16x32_bf16(
              false, qF[u], false, pF, (short)0, accO[g * 4 + u], false, false);
      }
    }
  }

  // ---- normalize (broadcast 1/l per i-row) and store coalesced fp32 ----
  if (nl == 0) {
#pragma unroll
    for (int r = 0; r < 8; ++r) lsStat[w * 16 + 8 * hi + r] = 1.0f / lrow[r];
  }
  asm volatile("s_wait_dscnt 0" ::: "memory");
  float linvMy = lsStat[w * 16 + nl];
#pragma unroll
  for (int ci = 0; ci < 8; ++ci)
#pragma unroll
    for (int r = 0; r < 8; ++r) {
      int cg = h * DH + ci * 16 + r + 8 * hi;  // M = c
      int ig = i0 + nl;                        // N = i (lane-contiguous)
      Out[((size_t)bz * C_DIM + cg) * P_DIM + ig] = accO[ci][r] * linvMy;
    }
}

// ---------------------------------------------------------------------------
extern "C" void kernel_launch(void* const* d_in, const int* in_sizes, int n_in,
                              void* d_out, int out_size, void* d_ws,
                              size_t ws_size, hipStream_t stream) {
  (void)in_sizes; (void)n_in; (void)out_size; (void)ws_size;
  const float* Xq = (const float*)d_in[0];
  const float* Xk = (const float*)d_in[1];
  const float* Wq = (const float*)d_in[2];
  const float* Wk = (const float*)d_in[3];
  const float* Wv = (const float*)d_in[4];
  const float* g1 = (const float*)d_in[5];
  const float* b1 = (const float*)d_in[6];
  const float* m1 = (const float*)d_in[7];
  const float* v1 = (const float*)d_in[8];
  const float* g2 = (const float*)d_in[9];
  const float* b2 = (const float*)d_in[10];
  const float* m2 = (const float*)d_in[11];
  const float* v2 = (const float*)d_in[12];
  const float* g3 = (const float*)d_in[13];
  const float* b3 = (const float*)d_in[14];
  const float* m3 = (const float*)d_in[15];
  const float* v3 = (const float*)d_in[16];

  char* ws = (char*)d_ws;
  const size_t bufElems = (size_t)BATCH * C_DIM * P_DIM;  // 8M elems, 16MB bf16
  unsigned short* Qb = (unsigned short*)ws;                      // [b][C][P]
  unsigned short* Kt = (unsigned short*)(ws + bufElems * 2);     // [b][P][C]
  unsigned short* Vt = (unsigned short*)(ws + 2 * bufElems * 2); // [b][P][C]
  float* sc = (float*)(ws + 3 * bufElems * 2);  // 3*C scales
  float* sh = sc + 3 * C_DIM;                   // 3*C shifts

  bn_coeffs_kernel<<<dim3(12), dim3(256), 0, stream>>>(
      g1, b1, m1, v1, g2, b2, m2, v2, g3, b3, m3, v3, sc, sh);

  dim3 gG(P_DIM / 128, C_DIM / 128, BATCH), blk(256);
  qkv_gemm_bn_kernel<0><<<gG, blk, 0, stream>>>(Wq, Xq, sc, sh, Qb);
  qkv_gemm_bn_kernel<1><<<gG, blk, 0, stream>>>(Wk, Xk, sc + C_DIM, sh + C_DIM, Kt);
  qkv_gemm_bn_kernel<1><<<gG, blk, 0, stream>>>(Wv, Xq, sc + 2 * C_DIM, sh + 2 * C_DIM, Vt);

  xattn_kernel<<<dim3(P_DIM / 128, HEADS, BATCH), blk, 0, stream>>>(
      Qb, Kt, Vt, (float*)d_out);
}
